// HeteroTransformerGNN_85976655331839
// MI455X (gfx1250) — matrix operand order repaired and checked by reading, not compiled, not run on hardware
//
#include <hip/hip_runtime.h>
#include <hip/hip_bf16.h>
#include <math.h>

// ---------------------------------------------------------------------------
// Types for CDNA5 WMMA
// ---------------------------------------------------------------------------
typedef __attribute__((ext_vector_type(16))) __bf16 v16bf;
typedef __attribute__((ext_vector_type(8)))  float  v8f;

union FragAB { v16bf v; uint4 u[2]; };

// fp32 -> bf16 (round to nearest even), stored as ushort
__device__ __forceinline__ unsigned short f2bf(float f) {
  unsigned u = __float_as_uint(f);
  unsigned r = ((u >> 16) & 1u) + 0x7FFFu;
  return (unsigned short)((u + r) >> 16);
}

// ---------------------------------------------------------------------------
// Convert kernels
// ---------------------------------------------------------------------------
__global__ __launch_bounds__(256) void cvt_f32_bf16(const float* __restrict__ in,
                                                    unsigned short* __restrict__ out,
                                                    size_t n) {
  size_t i = (size_t)blockIdx.x * blockDim.x + threadIdx.x;
  if (i < n) out[i] = f2bf(in[i]);
}

// W [K x N] fp32  ->  WT [N x K] bf16
__global__ __launch_bounds__(256) void wt_cvt(const float* __restrict__ W,
                                              unsigned short* __restrict__ WT,
                                              int K, int N) {
  size_t t = (size_t)blockIdx.x * blockDim.x + threadIdx.x;
  if (t >= (size_t)K * N) return;
  int k = (int)(t / N), n = (int)(t % N);
  WT[(size_t)n * K + k] = f2bf(W[t]);
}

// ---------------------------------------------------------------------------
// Direct-load WMMA GEMM (used for small K):
// C[M,N] = A[M,K](bf16) * BT[N,K]^T(bf16) + bias  (+= if accum)
// block = 256 threads = 8 waves; wave -> 32x32 tile; block -> 64x128
// ---------------------------------------------------------------------------
__global__ __launch_bounds__(256)
void wmma_gemm_bf16(const unsigned short* __restrict__ A,
                    const unsigned short* __restrict__ BT,
                    const float* __restrict__ bias,
                    float* __restrict__ C,
                    int M, int K, int N, int accum) {
  const int lane = threadIdx.x & 31;
  const int wave = threadIdx.x >> 5;
  const int wr   = wave & 1;
  const int wc   = wave >> 1;
  const int g    = lane >> 4;
  const int lm   = lane & 15;

  const int row0 = blockIdx.x * 64 + wr * 32;
  const int col0 = blockIdx.y * 128 + wc * 32;

  v8f acc[2][2] = {};

  const unsigned short* pa[2];
  const unsigned short* pb[2];
#pragma unroll
  for (int i = 0; i < 2; ++i) {
    int r = row0 + i * 16 + lm;
    if (r > M - 1) r = M - 1;     // clamp loads; stores are guarded
    pa[i] = A + (size_t)r * K;
  }
#pragma unroll
  for (int j = 0; j < 2; ++j) {
    int c = col0 + j * 16 + lm;
    pb[j] = BT + (size_t)c * K;
  }

  for (int k0 = 0; k0 < K; k0 += 32) {
    const int o0 = k0 + g * 8;
    const int o1 = k0 + 16 + g * 8;
    FragAB a[2], b[2];
#pragma unroll
    for (int i = 0; i < 2; ++i) {
      a[i].u[0] = *(const uint4*)(pa[i] + o0);
      a[i].u[1] = *(const uint4*)(pa[i] + o1);
      __builtin_prefetch(pa[i] + o0 + 64, 0, 3);
    }
#pragma unroll
    for (int j = 0; j < 2; ++j) {
      b[j].u[0] = *(const uint4*)(pb[j] + o0);
      b[j].u[1] = *(const uint4*)(pb[j] + o1);
      __builtin_prefetch(pb[j] + o0 + 64, 0, 3);
    }
#pragma unroll
    for (int i = 0; i < 2; ++i)
#pragma unroll
      for (int j = 0; j < 2; ++j)
        acc[i][j] = __builtin_amdgcn_wmma_f32_16x16x32_bf16(
            false, a[i].v, false, b[j].v, (short)0, acc[i][j], false, false);
  }

#pragma unroll
  for (int j = 0; j < 2; ++j) {
    const int col = col0 + j * 16 + lm;
    const float bv = bias ? bias[col] : 0.0f;
#pragma unroll
    for (int i = 0; i < 2; ++i) {
      const int rbase = row0 + i * 16 + g * 8;
#pragma unroll
      for (int r = 0; r < 8; ++r) {
        const int row = rbase + r;
        if (row < M) {
          size_t idx = (size_t)row * N + col;
          float v = acc[i][j][r] + bv;
          if (accum) C[idx] += v; else C[idx] = v;
        }
      }
    }
  }
}

// ---------------------------------------------------------------------------
// Async-copy + LDS double-buffered WMMA GEMM (used for K >= 512).
// Stage per k-step of 32: A tile 64x32 (4KB), B tile 128x32 (8KB), x2 buffers.
// Tiles are DMA'd with GLOBAL_LOAD_ASYNC_TO_LDS_B128 (ASYNCcnt) and consumed
// via ds_load_b128 fragment reads; DMA of stage i+1 overlaps WMMA of stage i.
// ---------------------------------------------------------------------------
__global__ __launch_bounds__(256)
void wmma_gemm_bf16_lds(const unsigned short* __restrict__ A,
                        const unsigned short* __restrict__ BT,
                        const float* __restrict__ bias,
                        float* __restrict__ C,
                        int M, int K, int N, int accum) {
  __shared__ unsigned short As[2][64 * 32];
  __shared__ unsigned short Bs[2][128 * 32];

  const int tid  = threadIdx.x;
  const int lane = tid & 31;
  const int wave = tid >> 5;
  const int wr   = wave & 1;
  const int wc   = wave >> 1;
  const int g    = lane >> 4;
  const int lm   = lane & 15;

  const int row0 = blockIdx.x * 64;
  const int col0 = blockIdx.y * 128;

  // --- async copy lane assignments ---------------------------------------
  // A tile: 64 rows x 32 k = 256 chunks of 16B; threads 0..255, 1 chunk each
  const int arow  = tid >> 2;
  const int apart = tid & 3;                 // 16B quarter of the 64B row
  int ar = row0 + arow; if (ar > M - 1) ar = M - 1;
  const unsigned short* agp = A + (size_t)ar * K + apart * 8; // 8 bf16 per chunk
  const unsigned aoff = (unsigned)(arow * 32 + apart * 8) * 2;   // bytes

  // B tile: 128 cols x 32 k = 512 chunks of 16B; 2 chunks per thread
  const int c0 = tid, c1 = tid + 256;
  const unsigned short* bgp0 = BT + (size_t)(col0 + (c0 >> 2)) * K + (c0 & 3) * 8;
  const unsigned short* bgp1 = BT + (size_t)(col0 + (c1 >> 2)) * K + (c1 & 3) * 8;
  const unsigned boff0 = (unsigned)((c0 >> 2) * 32 + (c0 & 3) * 8) * 2;
  const unsigned boff1 = (unsigned)((c1 >> 2) * 32 + (c1 & 3) * 8) * 2;

  // flat address of a __shared__ object keeps the LDS byte offset in [31:0]
  const unsigned ldsA0 = (unsigned)(uintptr_t)&As[0][0];
  const unsigned ldsA1 = (unsigned)(uintptr_t)&As[1][0];
  const unsigned ldsB0 = (unsigned)(uintptr_t)&Bs[0][0];
  const unsigned ldsB1 = (unsigned)(uintptr_t)&Bs[1][0];

  auto issue_stage = [&](int k0, int buf) {
    const unsigned la = (buf ? ldsA1 : ldsA0);
    const unsigned lb = (buf ? ldsB1 : ldsB0);
    {
      unsigned d = la + aoff;
      unsigned long long ga = (unsigned long long)(uintptr_t)(agp + k0);
      asm volatile("global_load_async_to_lds_b128 %0, %1, off"
                   :: "v"(d), "v"(ga) : "memory");
    }
    {
      unsigned d = lb + boff0;
      unsigned long long ga = (unsigned long long)(uintptr_t)(bgp0 + k0);
      asm volatile("global_load_async_to_lds_b128 %0, %1, off"
                   :: "v"(d), "v"(ga) : "memory");
    }
    {
      unsigned d = lb + boff1;
      unsigned long long ga = (unsigned long long)(uintptr_t)(bgp1 + k0);
      asm volatile("global_load_async_to_lds_b128 %0, %1, off"
                   :: "v"(d), "v"(ga) : "memory");
    }
  };

  v8f acc[2][2] = {};
  const int nIter = K >> 5;

  issue_stage(0, 0);
  for (int it = 0; it < nIter; ++it) {
    const int cur = it & 1;
    asm volatile("s_wait_asynccnt 0x0" ::: "memory");
    __syncthreads();                      // stage `it` visible to all waves
    if (it + 1 < nIter) issue_stage((it + 1) << 5, cur ^ 1);

    FragAB a[2], b[2];
#pragma unroll
    for (int i = 0; i < 2; ++i) {
      const unsigned short* p = &As[cur][(wr * 32 + i * 16 + lm) * 32];
      a[i].u[0] = *(const uint4*)(p + g * 8);
      a[i].u[1] = *(const uint4*)(p + 16 + g * 8);
    }
#pragma unroll
    for (int j = 0; j < 2; ++j) {
      const unsigned short* p = &Bs[cur][(wc * 32 + j * 16 + lm) * 32];
      b[j].u[0] = *(const uint4*)(p + g * 8);
      b[j].u[1] = *(const uint4*)(p + 16 + g * 8);
    }
#pragma unroll
    for (int i = 0; i < 2; ++i)
#pragma unroll
      for (int j = 0; j < 2; ++j)
        acc[i][j] = __builtin_amdgcn_wmma_f32_16x16x32_bf16(
            false, a[i].v, false, b[j].v, (short)0, acc[i][j], false, false);
    // no trailing barrier needed: buffer `cur` is only overwritten by the
    // issue in iteration it+1, which happens after that iteration's barrier
  }

#pragma unroll
  for (int j = 0; j < 2; ++j) {
    const int col = col0 + wc * 32 + j * 16 + lm;
    const float bv = bias ? bias[col] : 0.0f;
#pragma unroll
    for (int i = 0; i < 2; ++i) {
      const int rbase = row0 + wr * 32 + i * 16 + g * 8;
#pragma unroll
      for (int r = 0; r < 8; ++r) {
        const int row = rbase + r;
        if (row < M) {
          size_t idx = (size_t)row * N + col;
          float v = acc[i][j][r] + bv;
          if (accum) C[idx] += v; else C[idx] = v;
        }
      }
    }
  }
}

// ---------------------------------------------------------------------------
// Edge attention kernels (fp32, L2-resident gathers)
// ---------------------------------------------------------------------------
__global__ __launch_bounds__(256)
void edge_logits_max(const int* __restrict__ srcIdx, const int* __restrict__ dstIdx,
                     int E, const float* __restrict__ q, const float* __restrict__ k,
                     int Ch, int H, float scale,
                     float* __restrict__ logits, unsigned* __restrict__ segmax) {
  int t = blockIdx.x * blockDim.x + threadIdx.x;
  if (t >= E * H) return;
  int e = t / H, h = t % H;
  int s = srcIdx[e], d = dstIdx[e];
  const float* qp = q + (size_t)d * (H * Ch) + h * Ch;
  const float* kp = k + (size_t)s * (H * Ch) + h * Ch;
  float acc = 0.0f;
  for (int c = 0; c < Ch; ++c) acc += qp[c] * kp[c];
  acc *= scale;
  logits[t] = acc;
  unsigned key = __float_as_uint(acc);
  key = (key & 0x80000000u) ? ~key : (key | 0x80000000u);  // order-preserving
  atomicMax(&segmax[(size_t)d * H + h], key);
}

__global__ __launch_bounds__(256)
void edge_exp_sum(const int* __restrict__ dstIdx, int E, int H,
                  float* __restrict__ logits, const unsigned* __restrict__ segmax,
                  float* __restrict__ denom) {
  int t = blockIdx.x * blockDim.x + threadIdx.x;
  if (t >= E * H) return;
  int e = t / H, h = t % H;
  int d = dstIdx[e];
  unsigned key = segmax[(size_t)d * H + h];
  float m = (key & 0x80000000u) ? __uint_as_float(key ^ 0x80000000u)
                                : __uint_as_float(~key);
  float a = __expf(logits[t] - m);
  logits[t] = a;
  atomicAdd(&denom[(size_t)d * H + h], a);
}

__global__ __launch_bounds__(256)
void edge_scatter(const int* __restrict__ srcIdx, const int* __restrict__ dstIdx,
                  int E, const float* __restrict__ v, const float* __restrict__ a,
                  const float* __restrict__ denom, float* __restrict__ agg,
                  int H, int Ch) {
  const int HC = H * Ch;
  size_t t = (size_t)blockIdx.x * blockDim.x + threadIdx.x;
  if (t >= (size_t)E * HC) return;
  int e  = (int)(t / HC);
  int ch = (int)(t % HC);
  int h  = ch / Ch;
  int s = srcIdx[e], d = dstIdx[e];
  float alpha = a[(size_t)e * H + h] / (denom[(size_t)d * H + h] + 1e-16f);
  atomicAdd(&agg[(size_t)d * HC + ch], v[(size_t)s * HC + ch] * alpha);
}

// ---------------------------------------------------------------------------
// out = elu(LayerNorm(a + b)); one block per row, blockDim == Cdim (128/256)
// ---------------------------------------------------------------------------
__global__ __launch_bounds__(256)
void add_ln_elu(const float* __restrict__ a, const float* __restrict__ b,
                const float* __restrict__ gamma, const float* __restrict__ beta,
                float* __restrict__ out, int Cdim) {
  __shared__ float sh[256];
  const int row = blockIdx.x;
  const int c   = threadIdx.x;
  const size_t idx = (size_t)row * Cdim + c;
  float x = a[idx] + b[idx];
  sh[c] = x; __syncthreads();
  for (int s = Cdim >> 1; s > 0; s >>= 1) { if (c < s) sh[c] += sh[c + s]; __syncthreads(); }
  float mu = sh[0] / (float)Cdim; __syncthreads();
  float dx = x - mu;
  sh[c] = dx * dx; __syncthreads();
  for (int s = Cdim >> 1; s > 0; s >>= 1) { if (c < s) sh[c] += sh[c + s]; __syncthreads(); }
  float var = sh[0] / (float)Cdim;
  float y = dx * rsqrtf(var + 1e-5f) * gamma[c] + beta[c];
  out[idx] = (y > 0.0f) ? y : (__expf(y) - 1.0f);
}

// ---------------------------------------------------------------------------
// Host orchestration
// ---------------------------------------------------------------------------
extern "C" void kernel_launch(void* const* d_in, const int* in_sizes, int n_in,
                              void* d_out, int out_size, void* d_ws, size_t ws_size,
                              hipStream_t stream) {
  (void)out_size; (void)ws_size;
  if (n_in < 135) return;  // expected: 4 x + 128 param leaves + 3 edge arrays

  // sorted ntypes: 0=disease 1=drug 2=event 3=target
  static const int nrows[4]  = {5000, 10000, 50000, 10000};
  static const int rowoff[4] = {0, 5000, 15000, 65000};
  static const int xin[4]    = {2, 0, 3, 1};   // d_in index of x per sorted ntype
  const int TOT = 75000;

  // edge buffers are the last three inputs: drug_event, target_event, event_disease
  const int IE[3] = {n_in - 3, n_in - 2, n_in - 1};

  struct RelT { int src, dst, ebuf, srow, drow; };
  static const RelT rels[6] = {
    {0, 2, 2, 1, 0},  // disease__rev_treats__event
    {1, 2, 0, 0, 1},  // drug__participates__event
    {2, 1, 0, 1, 0},  // event__rev_participates__drug
    {2, 3, 1, 1, 0},  // event__rev_participates__target
    {2, 0, 2, 0, 1},  // event__treats__disease
    {3, 2, 1, 0, 1},  // target__participates__event
  };
  // per-rel param leaves (sorted): +0 Wk +1 Wq +2 Ws +3 Wv +4 bk +5 bq +6 bs +7 bv
  const int CONV_BASE[2] = {4, 52};
  const int LN_BASE[2]   = {100, 108};  // per ntype: +0 b, +1 g
  const int SK_BASE[2]   = {116, 124};  // per ntype: +0 W, +1 b

  // ---- workspace carve-up -------------------------------------------------
  char* wp = (char*)d_ws;
  auto alloc = [&](size_t bytes) -> void* {
    void* p = (void*)wp;
    wp += (bytes + 255) & ~(size_t)255;
    return p;
  };
  unsigned short* xb  = (unsigned short*)alloc((size_t)TOT * 1024 * 2);
  unsigned short* hb  = (unsigned short*)alloc((size_t)TOT * 256 * 2);
  float* agg1  = (float*)alloc((size_t)TOT * 256 * 4);
  float* sk1b  = (float*)alloc((size_t)TOT * 256 * 4);
  float* agg2  = (float*)alloc((size_t)TOT * 128 * 4);
  float* sk2b  = (float*)alloc((size_t)TOT * 128 * 4);
  float* qbuf  = (float*)alloc((size_t)50000 * 256 * 4);
  float* kbuf  = (float*)alloc((size_t)50000 * 256 * 4);
  float* vbuf  = (float*)alloc((size_t)50000 * 256 * 4);
  float* logb  = (float*)alloc((size_t)100000 * 8 * 4);
  unsigned* smax = (unsigned*)alloc((size_t)50000 * 8 * 4);
  float* den   = (float*)alloc((size_t)50000 * 8 * 4);

  unsigned short* cw[2][6][4];
  unsigned short* sw[2][4];
  {
    size_t o = 0;
    unsigned short* wb0 = (unsigned short*)alloc(
        (size_t)(6 * 4 * 1024 * 256 + 6 * 4 * 256 * 128 + 4 * 1024 * 256 + 4 * 256 * 128) * 2);
    for (int r = 0; r < 6; ++r) for (int m = 0; m < 4; ++m) { cw[0][r][m] = wb0 + o; o += 1024 * 256; }
    for (int r = 0; r < 6; ++r) for (int m = 0; m < 4; ++m) { cw[1][r][m] = wb0 + o; o += 256 * 128; }
    for (int t = 0; t < 4; ++t) { sw[0][t] = wb0 + o; o += 1024 * 256; }
    for (int t = 0; t < 4; ++t) { sw[1][t] = wb0 + o; o += 256 * 128; }
  }

  auto gemm = [&](const unsigned short* A, const unsigned short* BT, const float* bias,
                  float* Cp, int M, int K, int N, int accum) {
    dim3 grid((unsigned)((M + 63) / 64), (unsigned)(N / 128));
    if (K >= 512)
      wmma_gemm_bf16_lds<<<grid, 256, 0, stream>>>(A, BT, bias, Cp, M, K, N, accum);
    else
      wmma_gemm_bf16<<<grid, 256, 0, stream>>>(A, BT, bias, Cp, M, K, N, accum);
  };

  // ---- convert x to bf16 --------------------------------------------------
  for (int t = 0; t < 4; ++t) {
    size_t n = (size_t)nrows[t] * 1024;
    cvt_f32_bf16<<<(unsigned)((n + 255) / 256), 256, 0, stream>>>(
        (const float*)d_in[xin[t]], xb + (size_t)rowoff[t] * 1024, n);
  }
  // ---- convert + transpose all weights ------------------------------------
  for (int l = 0; l < 2; ++l) {
    int K = l ? 256 : 1024, N = l ? 128 : 256;
    unsigned grid = (unsigned)(((size_t)K * N + 255) / 256);
    for (int r = 0; r < 6; ++r)
      for (int m = 0; m < 4; ++m)
        wt_cvt<<<grid, 256, 0, stream>>>((const float*)d_in[CONV_BASE[l] + r * 8 + m],
                                         cw[l][r][m], K, N);
    for (int t = 0; t < 4; ++t)
      wt_cvt<<<grid, 256, 0, stream>>>((const float*)d_in[SK_BASE[l] + t * 2 + 0],
                                       sw[l][t], K, N);
  }

  // ---- two hetero layers --------------------------------------------------
  for (int l = 0; l < 2; ++l) {
    const int K  = l ? 256 : 1024;
    const int N  = l ? 128 : 256;
    const int Ch = N / 8;
    const float scale = 1.0f / sqrtf((float)Ch);
    const unsigned short* X = l ? hb : xb;
    const int xstride = K;
    float* agg = l ? agg2 : agg1;
    float* skb = l ? sk2b : sk1b;

    hipMemsetAsync(agg, 0, (size_t)TOT * N * 4, stream);

    for (int r = 0; r < 6; ++r) {
      const RelT& rl = rels[r];
      const int ebuf = IE[rl.ebuf];
      const int E = in_sizes[ebuf] / 2;
      const int* ebase = (const int*)d_in[ebuf];
      const int* sidx = ebase + (size_t)rl.srow * E;
      const int* didx = ebase + (size_t)rl.drow * E;
      const int Ms = nrows[rl.src], Md = nrows[rl.dst];
      const unsigned short* Xs = X + (size_t)rowoff[rl.src] * xstride;
      const unsigned short* Xd = X + (size_t)rowoff[rl.dst] * xstride;
      const int pb = CONV_BASE[l] + r * 8;

      gemm(Xd, cw[l][r][1], (const float*)d_in[pb + 5], qbuf, Md, K, N, 0);  // q
      gemm(Xs, cw[l][r][0], (const float*)d_in[pb + 4], kbuf, Ms, K, N, 0);  // k
      gemm(Xs, cw[l][r][3], (const float*)d_in[pb + 7], vbuf, Ms, K, N, 0);  // v
      gemm(Xd, cw[l][r][2], (const float*)d_in[pb + 6],
           agg + (size_t)rowoff[rl.dst] * N, Md, K, N, 1);                   // root/skip term

      hipMemsetAsync(smax, 0, (size_t)Md * 8 * 4, stream);
      hipMemsetAsync(den,  0, (size_t)Md * 8 * 4, stream);

      unsigned gEH = (unsigned)((E * 8 + 255) / 256);
      edge_logits_max<<<gEH, 256, 0, stream>>>(sidx, didx, E, qbuf, kbuf, Ch, 8, scale,
                                               logb, smax);
      edge_exp_sum<<<gEH, 256, 0, stream>>>(didx, E, 8, logb, smax, den);
      unsigned gEC = (unsigned)(((size_t)E * N + 255) / 256);
      edge_scatter<<<gEC, 256, 0, stream>>>(sidx, didx, E, vbuf, logb, den,
                                            agg + (size_t)rowoff[rl.dst] * N, 8, Ch);
    }

    // per-ntype skip projection
    for (int t = 0; t < 4; ++t)
      gemm(X + (size_t)rowoff[t] * xstride, sw[l][t],
           (const float*)d_in[SK_BASE[l] + t * 2 + 1],
           skb + (size_t)rowoff[t] * N, nrows[t], K, N, 0);

    // add + LayerNorm + ELU (layer1 in-place into agg1; layer2 -> d_out)
    float* outb = l ? (float*)d_out : agg1;
    for (int t = 0; t < 4; ++t) {
      const int lb = LN_BASE[l] + t * 2;
      add_ln_elu<<<(unsigned)nrows[t], (unsigned)N, 0, stream>>>(
          agg + (size_t)rowoff[t] * N, skb + (size_t)rowoff[t] * N,
          (const float*)d_in[lb + 1], (const float*)d_in[lb + 0],
          outb + (size_t)rowoff[t] * N, N);
    }

    if (!l) {
      size_t n = (size_t)TOT * 256;
      cvt_f32_bf16<<<(unsigned)((n + 255) / 256), 256, 0, stream>>>(agg1, hb, n);
    }
  }
}
